// Model_59889023975777
// MI455X (gfx1250) — matrix operand order repaired and checked by reading, not compiled, not run on hardware
//
#include <hip/hip_runtime.h>
#include <hip/hip_bf16.h>

// ---------------------------------------------------------------------------
// CDNA5 (gfx1250) GRU stack: bf16 WMMA (v_wmma_f32_16x16x32_bf16, wave32).
// Time-major activations [T, B, D]; weights used directly as [N][K] B-operand.
// Recurrence runs as ONE persistent kernel per layer (<=32 workgroups), with
// a device-scope atomic barrier per timestep instead of 512 kernel launches.
// Workspace: bf16 weights (~19 MB) | bf16 act (64 MB) | fp32 xg (402 MB) |
//            fp32 y (128 MB) | 6 barrier counters.
// ---------------------------------------------------------------------------

typedef __attribute__((ext_vector_type(16))) __bf16 v16bf;
typedef __attribute__((ext_vector_type(8)))  __bf16 v8bf;
typedef __attribute__((ext_vector_type(8)))  float  v8f;

union BF16Frag { v16bf v; v8bf h[2]; };

__device__ __forceinline__ __bf16 f2bf(float f) {
  unsigned u = __builtin_bit_cast(unsigned, f);
  u += 0x7FFFu + ((u >> 16) & 1u);                 // round-to-nearest-even
  unsigned short s = (unsigned short)(u >> 16);
  return __builtin_bit_cast(__bf16, s);
}

// -------------------------- fp32 -> bf16 convert ---------------------------
__global__ void k_cvt_bf16(const float* __restrict__ src,
                           __bf16* __restrict__ dst, int n) {
  int i = blockIdx.x * 256 + threadIdx.x;
  if (i < n) dst[i] = f2bf(src[i]);
}

// x [B,T,128] fp32 -> actBF [T,B,128] bf16 (time-major)
__global__ void k_cvt_x(const float* __restrict__ x, __bf16* __restrict__ dst) {
  int i = blockIdx.x * 256 + threadIdx.x;          // < 64*512*128
  int c = i & 127;
  int t = (i >> 7) & 511;
  int b = i >> 16;
  dst[(size_t)((t << 6) + b) * 128 + c] = f2bf(x[i]);
}

// Zero the 6 per-layer grid-barrier counters (one 256B slot each).
__global__ void k_init_counters(unsigned* __restrict__ c) {
  if (threadIdx.x < 6) c[threadIdx.x * 64] = 0u;
}

// ----------------------- input-projection GEMM -----------------------------
// C[M,N] = A[M,K](bf16) * W[N,K](bf16)^T, fp32 accum.  Block tile 128x128,
// 8 waves (4 along M x 2 along N), each wave: 2x4 fragments of 16x16.
#define LSTR 40   // LDS row stride in bf16 elems (80 B: 16B-aligned, bank-spread)

__global__ __launch_bounds__(256) void k_gemm_xg(
    const __bf16* __restrict__ A, const __bf16* __restrict__ W,
    float* __restrict__ C, int K, int N) {
  __shared__ __align__(16) __bf16 sA[128 * LSTR];
  __shared__ __align__(16) __bf16 sB[128 * LSTR];

  const int tid  = threadIdx.x;
  const int lane = tid & 31;
  const int wid  = tid >> 5;
  const int wm   = wid & 3;                // wave M position (0..3)
  const int wn   = wid >> 2;               // wave N position (0..1)
  const int mBase = blockIdx.y * 128;
  const int nBase = blockIdx.x * 128;

  const int lq = lane & 15;
  const int lh = lane >> 4;                // half-wave select
  const int aK0 = lh * 8, aK1 = 16 + lh * 8;   // A frag K chunks (ISA 7.12.2)
  const int bK0 = lh * 8;                      // B frag K chunks (same as A)

  const int ldRow  = tid >> 1;             // 0..127
  const int ldHalf = (tid & 1) * 16;       // element offset 0 / 16

  v8f acc[2][4] = {};

  for (int k0 = 0; k0 < K; k0 += 32) {
    // Cooperative tile loads: 32 B per thread per tile (2 x b128).
    const uint4* gA = (const uint4*)(A + (size_t)(mBase + ldRow) * K + k0 + ldHalf);
    const uint4* gB = (const uint4*)(W + (size_t)(nBase + ldRow) * K + k0 + ldHalf);
    *(uint4*)(&sA[ldRow * LSTR + ldHalf])     = gA[0];
    *(uint4*)(&sA[ldRow * LSTR + ldHalf + 8]) = gA[1];
    *(uint4*)(&sB[ldRow * LSTR + ldHalf])     = gB[0];
    *(uint4*)(&sB[ldRow * LSTR + ldHalf + 8]) = gB[1];
    __syncthreads();

    BF16Frag af[2], bf[4];
#pragma unroll
    for (int mf = 0; mf < 2; ++mf) {
      const __bf16* p = &sA[(wm * 32 + mf * 16 + lq) * LSTR];
      af[mf].h[0] = *(const v8bf*)(p + aK0);
      af[mf].h[1] = *(const v8bf*)(p + aK1);
    }
#pragma unroll
    for (int nf = 0; nf < 4; ++nf) {
      const __bf16* p = &sB[(wn * 64 + nf * 16 + lq) * LSTR + bK0];
      bf[nf].h[0] = *(const v8bf*)(p);
      bf[nf].h[1] = *(const v8bf*)(p + 16);
    }
#pragma unroll
    for (int mf = 0; mf < 2; ++mf)
#pragma unroll
      for (int nf = 0; nf < 4; ++nf)
        acc[mf][nf] = __builtin_amdgcn_wmma_f32_16x16x32_bf16(
            false, af[mf].v, false, bf[nf].v, (short)0, acc[mf][nf],
            false, false);
    __syncthreads();
  }

#pragma unroll
  for (int mf = 0; mf < 2; ++mf) {
#pragma unroll
    for (int nf = 0; nf < 4; ++nf) {
      const int m0 = mBase + wm * 32 + mf * 16 + 8 * lh;   // C layout: M=i+8*lh
      const int n  = nBase + wn * 64 + nf * 16 + lq;
#pragma unroll
      for (int i = 0; i < 8; ++i)
        C[(size_t)(m0 + i) * N + n] = acc[mf][nf][i];
    }
  }
}

// ------------------- persistent GRU recurrence (per layer) -----------------
// Grid: H/32 blocks (<=32, all co-resident), 256 threads = 8 waves.
// Each block owns a 32-column hidden slice; each wave one (mf, nf) 16x16 tile
// with three fp32 gate accumulators (r/z/n).  All 512 timesteps run inside
// the kernel; blocks synchronize per step with a device-scope atomic barrier.
__global__ __launch_bounds__(256) void k_gru_persist(
    const float* __restrict__ xg,    // [T*B, 3H]
    const __bf16* __restrict__ Whh,  // [3H, H]
    float* __restrict__ y,           // [T*B, H]
    int H, unsigned* __restrict__ cnt) {
  __shared__ __align__(16) __bf16 sA[64 * LSTR];

  const int tid  = threadIdx.x;
  const int lane = tid & 31;
  const int wid  = tid >> 5;
  const int mf   = wid & 3;                // rows mf*16..mf*16+15
  const int nf   = wid >> 2;               // cols nf*16 within 32-col slice
  const int js   = blockIdx.x * 32;
  const int N3   = 3 * H;
  const int lq = lane & 15, lh = lane >> 4;

  const int ldRow = tid >> 2;              // 0..63
  const int ldSeg = (tid & 3) * 8;         // 0,8,16,24

  const int nCol = js + nf * 16 + lq;      // B-operand output column
  const int m0   = mf * 16 + 8 * lh;       // C layout base row
  const unsigned nb = gridDim.x;

  for (int t = 0; t < 512; ++t) {
    const float* yprev = y + ((size_t)t - 1) * 64 * H;

    // Prefetch this tile's xg gate rows into cache under the WMMA loop.
    {
      const float* px = xg + ((size_t)t * 64 + m0) * N3 + nCol;
      __builtin_prefetch(px, 0, 1);
      __builtin_prefetch(px + H, 0, 1);
      __builtin_prefetch(px + 2 * H, 0, 1);
    }

    v8f accR = {}, accZ = {}, accN = {};

    for (int k0 = 0; k0 < H; k0 += 32) {
      // Build bf16 A tile of h_{t-1} (zeros at t==0).
      v8bf pack = {};
      if (t > 0) {
        const float4* p = (const float4*)(yprev + (size_t)ldRow * H + k0 + ldSeg);
        float4 f0 = p[0], f1 = p[1];
        float fv[8] = {f0.x, f0.y, f0.z, f0.w, f1.x, f1.y, f1.z, f1.w};
#pragma unroll
        for (int j = 0; j < 8; ++j) pack[j] = f2bf(fv[j]);
      }
      *(v8bf*)(&sA[ldRow * LSTR + ldSeg]) = pack;
      __syncthreads();

      BF16Frag af;
      const __bf16* pa = &sA[(mf * 16 + lq) * LSTR];
      af.h[0] = *(const v8bf*)(pa + lh * 8);
      af.h[1] = *(const v8bf*)(pa + 16 + lh * 8);

#pragma unroll
      for (int g = 0; g < 3; ++g) {
        BF16Frag bf;
        const __bf16* pb = Whh + (size_t)(g * H + nCol) * H + k0 + lh * 8;
        bf.h[0] = *(const v8bf*)(pb);
        bf.h[1] = *(const v8bf*)(pb + 16);
        v8f* acc = (g == 0) ? &accR : (g == 1) ? &accZ : &accN;
        *acc = __builtin_amdgcn_wmma_f32_16x16x32_bf16(
            false, af.v, false, bf.v, (short)0, *acc, false, false);
      }
      __syncthreads();
    }

    // Gate math in fp32 on the C-layout (M = i + 8*lh, N = lq).
#pragma unroll
    for (int i = 0; i < 8; ++i) {
      const int m = m0 + i;
      const size_t row = (size_t)t * 64 + m;
      float xr = xg[row * N3 + nCol];
      float xz = xg[row * N3 + H + nCol];
      float xn = xg[row * N3 + 2 * H + nCol];
      float hp = (t > 0) ? yprev[(size_t)m * H + nCol] : 0.0f;
      float r  = 1.0f / (1.0f + __expf(-(xr + accR[i])));
      float z  = 1.0f / (1.0f + __expf(-(xz + accZ[i])));
      float nn = tanhf(xn + r * accN[i]);
      y[row * H + nCol] = (1.0f - z) * nn + z * hp;
    }

    // ---- device-scope grid barrier: all blocks finished step t ----
    __threadfence();                       // release this block's y[t] stores
    __syncthreads();
    if (tid == 0) {
      __hip_atomic_fetch_add(cnt, 1u, __ATOMIC_RELEASE,
                             __HIP_MEMORY_SCOPE_AGENT);
      const unsigned target = nb * (unsigned)(t + 1);
      while (__hip_atomic_load(cnt, __ATOMIC_ACQUIRE,
                               __HIP_MEMORY_SCOPE_AGENT) < target)
        __builtin_amdgcn_s_sleep(1);
    }
    __syncthreads();
    __threadfence();                       // acquire other blocks' y[t]
  }
}

// -------------------------------- LayerNorm --------------------------------
// One block per row (t*64+b). outBF: bf16 next-layer input (time-major).
// outF (last layer): fp32 d_out in [B,T,H] order.
__global__ __launch_bounds__(256) void k_layernorm(
    const float* __restrict__ y, const float* __restrict__ gw,
    const float* __restrict__ bw, int H,
    __bf16* __restrict__ outBF, float* __restrict__ outF) {
  __shared__ float sSum[256];
  __shared__ float sSq[256];
  const int r   = blockIdx.x;
  const int tid = threadIdx.x;
  const float* row = y + (size_t)r * H;

  float vals[4];
  float s = 0.f, sq = 0.f;
  int cnt = 0;
  for (int c = tid; c < H; c += 256) {
    float v = row[c];
    vals[cnt++] = v;
    s += v; sq += v * v;
  }
  sSum[tid] = s; sSq[tid] = sq;
  __syncthreads();
  for (int st = 128; st > 0; st >>= 1) {
    if (tid < st) { sSum[tid] += sSum[tid + st]; sSq[tid] += sSq[tid + st]; }
    __syncthreads();
  }
  const float mean = sSum[0] / (float)H;
  const float var  = sSq[0] / (float)H - mean * mean;
  const float inv  = rsqrtf(var);                    // eps = 0 per reference

  cnt = 0;
  for (int c = tid; c < H; c += 256) {
    float o = (vals[cnt++] - mean) * inv * gw[c] + bw[c];
    if (outBF) {
      outBF[(size_t)r * H + c] = f2bf(o);
    } else {
      const int tt = r >> 6, bb = r & 63;            // r = t*64 + b
      outF[((size_t)bb * 512 + tt) * H + c] = o;
    }
  }
}

// --------------------------------- driver ----------------------------------
extern "C" void kernel_launch(void* const* d_in, const int* in_sizes, int n_in,
                              void* d_out, int out_size, void* d_ws, size_t ws_size,
                              hipStream_t stream) {
  (void)in_sizes; (void)n_in; (void)out_size; (void)ws_size;
  static const int DIN[6] = {128, 256, 512, 1024, 512, 256};
  static const int HS [6] = {256, 512, 1024, 512, 256, 128};
  const int M = 64 * 512;                   // tokens

  char* ws = (char*)d_ws;
  size_t off = 0;
  auto carve = [&](size_t bytes) -> void* {
    void* p = ws + off;
    off = (off + bytes + 255) & ~(size_t)255;
    return p;
  };
  __bf16* wih[6]; __bf16* whh[6];
  for (int l = 0; l < 6; ++l) {
    wih[l] = (__bf16*)carve((size_t)3 * HS[l] * DIN[l] * sizeof(__bf16));
    whh[l] = (__bf16*)carve((size_t)3 * HS[l] * HS[l]  * sizeof(__bf16));
  }
  __bf16*   actBF = (__bf16*)carve((size_t)M * 1024 * sizeof(__bf16));
  float*    xg    = (float*) carve((size_t)M * 3072 * sizeof(float));
  float*    yb    = (float*) carve((size_t)M * 1024 * sizeof(float));
  unsigned* cnts  = (unsigned*)carve(6 * 64 * sizeof(unsigned)); // 256B/slot

  // Per-replay deterministic init of the grid-barrier counters.
  k_init_counters<<<1, 64, 0, stream>>>(cnts);

  // Weights -> bf16 once per call (L2-resident thereafter: ~19 MB vs 192 MB).
  for (int l = 0; l < 6; ++l) {
    const int nIh = 3 * HS[l] * DIN[l], nHh = 3 * HS[l] * HS[l];
    k_cvt_bf16<<<(nIh + 255) / 256, 256, 0, stream>>>(
        (const float*)d_in[1 + 4 * l], wih[l], nIh);
    k_cvt_bf16<<<(nHh + 255) / 256, 256, 0, stream>>>(
        (const float*)d_in[2 + 4 * l], whh[l], nHh);
  }
  // x [B,T,I] -> time-major bf16 [T,B,I].
  k_cvt_x<<<(64 * 512 * 128) / 256, 256, 0, stream>>>(
      (const float*)d_in[0], actBF);

  for (int l = 0; l < 6; ++l) {
    const int K = DIN[l], H = HS[l], N3 = 3 * H;
    // Input projections: one big WMMA GEMM.
    k_gemm_xg<<<dim3(N3 / 128, M / 128), 256, 0, stream>>>(
        actBF, wih[l], xg, K, N3);
    // Persistent recurrence: all 512 steps in one kernel, grid barrier/step.
    k_gru_persist<<<dim3(H / 32), 256, 0, stream>>>(
        xg, whh[l], yb, H, cnts + l * 64);
    // LayerNorm -> next-layer bf16 input, or final fp32 output.
    const bool last = (l == 5);
    k_layernorm<<<dim3(M), 256, 0, stream>>>(
        yb, (const float*)d_in[3 + 4 * l], (const float*)d_in[4 + 4 * l], H,
        last ? nullptr : actBF, last ? (float*)d_out : nullptr);
  }
}
